// COMPER_24464133718492
// MI455X (gfx1250) — compile-verified, matrix operand order checked by reading, not run on hardware
//
#include <hip/hip_runtime.h>
#include <hip/hip_bf16.h>

// ---------------------------------------------------------------------------
// COMPER forward on gfx1250: fused gather + 3-step LSTM + attention via
// v_wmma_f32_16x16x32_f16, then a small per-batch softmax/aggregate/MLP head.
// ---------------------------------------------------------------------------

typedef __attribute__((ext_vector_type(16))) _Float16 v16h;
typedef __attribute__((ext_vector_type(8)))  _Float16 v8h;
typedef __attribute__((ext_vector_type(8)))  float    v8f;

#define CB   8192      // batch
#define CP   10        // paths
#define CL   3         // path length
#define CE   128       // embedding dim
#define CH   128       // hidden
#define CF   384       // 3*E  (LSTM input / attention input width)
#define CG   512       // 4*H  (gate width)
#define CBP  (CB*CP)   // 81920 sequences
#define TILE_M 16

static __device__ __forceinline__ v16h cat8(v8h lo, v8h hi) {
    return __builtin_shufflevector(lo, hi, 0,1,2,3,4,5,6,7,8,9,10,11,12,13,14,15);
}
static __device__ __forceinline__ v8f splat8(float x) {
    v8f r = {x,x,x,x,x,x,x,x};
    return r;
}
static __device__ __forceinline__ float sigmoidf_(float x) {
    return 1.0f / (1.0f + __expf(-x));
}

// ---------------------------------------------------------------------------
// Kernel 0: pack weights to f16 (and transpose att_w1), fuse LSTM biases.
// ---------------------------------------------------------------------------
__global__ __launch_bounds__(256) void comper_pack_kernel(
    const float* __restrict__ Wih, const float* __restrict__ Whh,
    const float* __restrict__ bih, const float* __restrict__ bhh,
    const float* __restrict__ aw1,
    _Float16* __restrict__ Wih16, _Float16* __restrict__ Whh16,
    _Float16* __restrict__ Aw1t16, float* __restrict__ bias)
{
    int i = blockIdx.x * blockDim.x + threadIdx.x;
    if (i < CG * CF)  Wih16[i] = (_Float16)Wih[i];                 // [512][384]
    if (i < CG * CH)  Whh16[i] = (_Float16)Whh[i];                 // [512][128]
    if (i < 64 * CF) {                                             // att_w1^T: [64][384]
        int n = i / CF, k = i % CF;
        Aw1t16[i] = (_Float16)aw1[k * 64 + n];
    }
    if (i < CG)       bias[i] = bih[i] + bhh[i];
}

// ---------------------------------------------------------------------------
// Kernel 1: per 16-sequence tile — gather features, 3-step LSTM (WMMA),
// attention MLP hidden (WMMA) + score. Writes h_last (path_embs) and scores.
// ---------------------------------------------------------------------------
__global__ __launch_bounds__(256) void comper_lstm_att_kernel(
    const int*   __restrict__ user,  const int* __restrict__ item,
    const int*   __restrict__ pnode, const int* __restrict__ ptype,
    const int*   __restrict__ prel,
    const float* __restrict__ uemb,  const float* __restrict__ iemb,
    const float* __restrict__ remb,  const float* __restrict__ temb,
    const _Float16* __restrict__ Wih16, const _Float16* __restrict__ Whh16,
    const float* __restrict__ bias,
    const _Float16* __restrict__ Aw1t16,
    const float* __restrict__ att_b1, const float* __restrict__ att_w2,
    const float* __restrict__ att_b2,
    float* __restrict__ scores, float* __restrict__ pathemb)
{
    __shared__ alignas(16) _Float16 sX[TILE_M][CF];     // 12 KB: features / att input
    __shared__ alignas(16) float    sGates[TILE_M][CG]; // 32 KB: gate pre-activations
    __shared__ alignas(16) float    sC[TILE_M][CH];     //  8 KB: cell state
    __shared__ alignas(16) _Float16 sH[TILE_M][CH];     //  4 KB: hidden (f16, next A)
    __shared__ int sNode[CL][TILE_M], sType[CL][TILE_M], sRel[CL][TILE_M];

    const int tid  = threadIdx.x;
    const int wave = tid >> 5;
    const int lane = tid & 31;
    const int ln16 = lane & 15;
    const int off  = (lane >> 4) * 8;     // hi-half lanes take k-offset +8
    const int rbase = (lane >> 4) * 8;    // D rows: VGPR r -> M = r + 8*(lane>=16)
    const int s0   = blockIdx.x * TILE_M; // first sequence of this tile

    // Load per-step indices for the 16 sequences.
    if (tid < CL * TILE_M) {
        int t = tid / TILE_M, m = tid % TILE_M;
        int s = s0 + m;
        sNode[t][m] = pnode[s * CL + t];
        sType[t][m] = ptype[s * CL + t];
        sRel[t][m]  = prel[s * CL + t];
    }
    for (int e = tid; e < TILE_M * CH; e += 256) {
        int m = e >> 7, j = e & 127;
        sC[m][j] = 0.0f;
        sH[m][j] = (_Float16)0.0f;
    }
    __syncthreads();

    // ---------------- LSTM over L=3 steps ----------------
    for (int t = 0; t < CL; ++t) {
        // Gather [node_emb | type_emb | rel_emb] -> sX (f32 -> f16).
        for (int e = tid; e < TILE_M * CF; e += 256) {
            int m = e / CF, c = e % CF;
            float v;
            if (c < CE) {
                int node = sNode[t][m];
                v = (sType[t][m] == 0) ? uemb[node * CE + c] : iemb[node * CE + c];
            } else if (c < 2 * CE) {
                v = temb[sType[t][m] * CE + (c - CE)];
            } else {
                v = remb[sRel[t][m] * CE + (c - 2 * CE)];
            }
            sX[m][c] = (_Float16)v;
        }
        __syncthreads();

        // gates = X @ Wih^T + h @ Whh^T + bias, via WMMA.
        // Wave w owns gate columns [64w, 64w+64): 4 N-tiles of 16.
        const int n0base = wave * 64;
        v8f acc[4];
        for (int j = 0; j < 4; ++j)
            acc[j] = splat8(bias[n0base + j * 16 + ln16]);

        const _Float16* xr = &sX[ln16][0];
        for (int kk = 0; kk < CF / 32; ++kk) {
            int k0 = kk * 32;
            v16h a = cat8(*(const v8h*)(xr + k0 + off),
                          *(const v8h*)(xr + k0 + 16 + off));
            for (int j = 0; j < 4; ++j) {
                const _Float16* wr = Wih16 + (size_t)(n0base + j * 16 + ln16) * CF;
                v16h bmat = cat8(*(const v8h*)(wr + k0 + off),
                                 *(const v8h*)(wr + k0 + 16 + off));
                acc[j] = __builtin_amdgcn_wmma_f32_16x16x32_f16(
                    false, a, false, bmat, (short)0, acc[j], false, false);
            }
        }
        if (t > 0) {  // h == 0 on the first step
            const _Float16* hr = &sH[ln16][0];
            for (int kk = 0; kk < CH / 32; ++kk) {
                int k0 = kk * 32;
                v16h a = cat8(*(const v8h*)(hr + k0 + off),
                              *(const v8h*)(hr + k0 + 16 + off));
                for (int j = 0; j < 4; ++j) {
                    const _Float16* wr = Whh16 + (size_t)(n0base + j * 16 + ln16) * CH;
                    v16h bmat = cat8(*(const v8h*)(wr + k0 + off),
                                     *(const v8h*)(wr + k0 + 16 + off));
                    acc[j] = __builtin_amdgcn_wmma_f32_16x16x32_f16(
                        false, a, false, bmat, (short)0, acc[j], false, false);
                }
            }
        }
        // Spill gates to LDS (D layout: VGPR r, lane l -> M=r+8*(l>=16), N=n0+l%16).
        for (int j = 0; j < 4; ++j) {
            int n = n0base + j * 16 + ln16;
            for (int r = 0; r < 8; ++r)
                sGates[rbase + r][n] = acc[j][r];
        }
        __syncthreads();

        // Elementwise LSTM update (f32), torch gate order (i,f,g,o).
        for (int e = tid; e < TILE_M * CH; e += 256) {
            int m = e >> 7, j = e & 127;
            float ig = sGates[m][j];
            float fg = sGates[m][j + CH];
            float gg = sGates[m][j + 2 * CH];
            float og = sGates[m][j + 3 * CH];
            float c  = sigmoidf_(fg) * sC[m][j] + sigmoidf_(ig) * tanhf(gg);
            float h  = sigmoidf_(og) * tanhf(c);
            sC[m][j] = c;
            sH[m][j] = (_Float16)h;
        }
        __syncthreads();
    }

    // Emit path embeddings (h_last) for the aggregation kernel.
    for (int e = tid; e < TILE_M * CH; e += 256) {
        int m = e >> 7, j = e & 127;
        pathemb[(size_t)(s0 + m) * CH + j] = (float)sH[m][j];
    }

    // Build attention input [h | user_emb[user] | item_emb[item]] in sX.
    for (int e = tid; e < TILE_M * CF; e += 256) {
        int m = e / CF, c = e % CF;
        int b = (s0 + m) / CP;
        _Float16 v;
        if      (c < CE)      v = sH[m][c];
        else if (c < 2 * CE)  v = (_Float16)uemb[user[b] * CE + (c - CE)];
        else                  v = (_Float16)iemb[item[b] * CE + (c - 2 * CE)];
        sX[m][c] = v;
    }
    __syncthreads();

    // Attention hidden (16 x 64) = relu(att_in @ att_w1 + b1), waves 0..3.
    if (wave < 4) {
        const int n0 = wave * 16;
        v8f acc = splat8(att_b1[n0 + ln16]);
        const _Float16* xr = &sX[ln16][0];
        for (int kk = 0; kk < CF / 32; ++kk) {
            int k0 = kk * 32;
            v16h a = cat8(*(const v8h*)(xr + k0 + off),
                          *(const v8h*)(xr + k0 + 16 + off));
            const _Float16* wr = Aw1t16 + (size_t)(n0 + ln16) * CF;
            v16h bmat = cat8(*(const v8h*)(wr + k0 + off),
                             *(const v8h*)(wr + k0 + 16 + off));
            acc = __builtin_amdgcn_wmma_f32_16x16x32_f16(
                false, a, false, bmat, (short)0, acc, false, false);
        }
        for (int r = 0; r < 8; ++r) {
            float v = acc[r];
            sGates[rbase + r][n0 + ln16] = v > 0.0f ? v : 0.0f;   // relu
        }
    }
    __syncthreads();

    // Score: hidden(64) . att_w2 + b2, one row per thread.
    if (tid < TILE_M) {
        float s = att_b2[0];
        for (int k = 0; k < 64; ++k) s += sGates[tid][k] * att_w2[k];
        scores[s0 + tid] = s;
    }
}

// ---------------------------------------------------------------------------
// Kernel 2: per-batch softmax over paths, weighted aggregation, output MLP.
// d_out layout: pred [B] then weights [B*P].
// ---------------------------------------------------------------------------
__global__ __launch_bounds__(128) void comper_head_kernel(
    const float* __restrict__ scores, const float* __restrict__ pathemb,
    const float* __restrict__ fcw1, const float* __restrict__ fcb1,
    const float* __restrict__ fcw2, const float* __restrict__ fcb2,
    float* __restrict__ out)
{
    __shared__ float w[CP];
    __shared__ float agg[CH];
    __shared__ float hid[32];

    const int b = blockIdx.x, tid = threadIdx.x;

    if (tid == 0) {
        float sc[CP], mx = -1e30f;
        for (int p = 0; p < CP; ++p) { sc[p] = scores[b * CP + p]; mx = fmaxf(mx, sc[p]); }
        float sum = 0.0f;
        for (int p = 0; p < CP; ++p) { float e = __expf(sc[p] - mx); w[p] = e; sum += e; }
        float inv = 1.0f / sum;
        for (int p = 0; p < CP; ++p) w[p] *= inv;
    }
    __syncthreads();
    if (tid < CP) out[CB + b * CP + tid] = w[tid];

    float a = 0.0f;
    for (int p = 0; p < CP; ++p)
        a += w[p] * pathemb[((size_t)b * CP + p) * CH + tid];
    agg[tid] = a;
    __syncthreads();

    if (tid < 32) {
        float h = fcb1[tid];
        for (int j = 0; j < CH; ++j) h += agg[j] * fcw1[j * 32 + tid];
        hid[tid] = h > 0.0f ? h : 0.0f;
    }
    __syncthreads();

    if (tid == 0) {
        float p = fcb2[0];
        for (int k = 0; k < 32; ++k) p += hid[k] * fcw2[k];
        out[b] = p;
    }
}

// ---------------------------------------------------------------------------
extern "C" void kernel_launch(void* const* d_in, const int* in_sizes, int n_in,
                              void* d_out, int out_size, void* d_ws, size_t ws_size,
                              hipStream_t stream) {
    (void)in_sizes; (void)n_in; (void)out_size; (void)ws_size;

    const int*   user   = (const int*)  d_in[0];
    const int*   item   = (const int*)  d_in[1];
    const int*   pnode  = (const int*)  d_in[2];
    const int*   ptype  = (const int*)  d_in[3];
    const int*   prel   = (const int*)  d_in[4];
    const float* uemb   = (const float*)d_in[5];
    const float* iemb   = (const float*)d_in[6];
    const float* remb   = (const float*)d_in[7];
    const float* temb   = (const float*)d_in[8];
    const float* Wih    = (const float*)d_in[9];
    const float* Whh    = (const float*)d_in[10];
    const float* bih    = (const float*)d_in[11];
    const float* bhh    = (const float*)d_in[12];
    const float* aw1    = (const float*)d_in[13];
    const float* ab1    = (const float*)d_in[14];
    const float* aw2    = (const float*)d_in[15];
    const float* ab2    = (const float*)d_in[16];
    const float* fcw1   = (const float*)d_in[17];
    const float* fcb1   = (const float*)d_in[18];
    const float* fcw2   = (const float*)d_in[19];
    const float* fcb2   = (const float*)d_in[20];

    // Workspace layout (all offsets 256B-aligned).
    char* ws = (char*)d_ws;
    size_t o = 0;
    _Float16* Wih16  = (_Float16*)(ws + o); o += (size_t)CG * CF * 2;   // 384 KB
    _Float16* Whh16  = (_Float16*)(ws + o); o += (size_t)CG * CH * 2;   // 128 KB
    _Float16* Aw1t16 = (_Float16*)(ws + o); o += (size_t)64 * CF * 2;   //  48 KB
    float*    bias   = (float*)   (ws + o); o += (size_t)CG * 4;        //   2 KB
    float*    scores = (float*)   (ws + o); o += (size_t)CBP * 4;       // 320 KB
    float*    pathe  = (float*)   (ws + o); o += (size_t)CBP * CH * 4;  //  40 MB

    float* out = (float*)d_out;

    comper_pack_kernel<<<(CG * CF + 255) / 256, 256, 0, stream>>>(
        Wih, Whh, bih, bhh, aw1, Wih16, Whh16, Aw1t16, bias);

    comper_lstm_att_kernel<<<CBP / TILE_M, 256, 0, stream>>>(
        user, item, pnode, ptype, prel,
        uemb, iemb, remb, temb,
        Wih16, Whh16, bias, Aw1t16, ab1, aw2, ab2,
        scores, pathe);

    comper_head_kernel<<<CB, 128, 0, stream>>>(
        scores, pathe, fcw1, fcb1, fcw2, fcb2, out);
}
